// MultiheadedAttentionFM_90022514524806
// MI455X (gfx1250) — matrix operand order repaired and checked by reading, not compiled, not run on hardware
//
#include <hip/hip_runtime.h>

typedef __attribute__((ext_vector_type(16))) _Float16 v16h;
typedef __attribute__((ext_vector_type(8)))  _Float16 v8h;
typedef __attribute__((ext_vector_type(8)))  float    v8f;
typedef __attribute__((ext_vector_type(4)))  int      v4i;

union FragU { v16h v; v8h h[2]; };

// ---------------------------------------------------------------------------
// CDNA5 async global->LDS copy (16B granule), with sync fallback.
// Builtin signature (from hipcc diagnostic): (AS1 v4i*, AS3 v4i*, imm, imm).
// ---------------------------------------------------------------------------
#if __has_builtin(__builtin_amdgcn_global_load_async_to_lds_b128)
#define HAVE_ASYNC_LDS 1
#else
#define HAVE_ASYNC_LDS 0
#endif

__device__ inline void async_copy16B(const _Float16* g, _Float16* l) {
#if HAVE_ASYNC_LDS
  __builtin_amdgcn_global_load_async_to_lds_b128(
      (__attribute__((address_space(1))) v4i*)g,
      (__attribute__((address_space(3))) v4i*)l, 0, 0);
#else
  *(v8h*)l = *(const v8h*)g;
#endif
}

__device__ inline void wait_async_copies() {
#if HAVE_ASYNC_LDS
#if __has_builtin(__builtin_amdgcn_s_wait_asynccnt)
  __builtin_amdgcn_s_wait_asynccnt(0);
#else
  asm volatile("s_wait_asynccnt 0x0" ::: "memory");
#endif
#endif
}

// ---------------------------------------------------------------------------
// WMMA helpers (CDNA5 16x16x32 f16 -> f32)
// ---------------------------------------------------------------------------
__device__ inline v8f wmma_f16(v16h a, v16h b, v8f c) {
  return __builtin_amdgcn_wmma_f32_16x16x32_f16(false, a, false, b, (short)0, c,
                                                false, false);
}

// A fragment: 16x32 f16, row-major tile, leading dim `ld` (halfs).
__device__ inline v16h load_frag_a(const _Float16* base, int ld, int lane) {
  const int m = lane & 15, hh = lane >> 4;
  const _Float16* p = base + m * ld;
  FragU f;
  f.h[0] = *(const v8h*)(p + hh * 8);
  f.h[1] = *(const v8h*)(p + 16 + hh * 8);
  return f.v;
}

// B fragment: 32x16 f16 tile supplied TRANSPOSED (Bt[n][k], leading dim ld).
__device__ inline v16h load_frag_bt(const _Float16* base, int ld, int lane) {
  const int n = lane & 15, hh = lane >> 4;
  const _Float16* p = base + n * ld + hh * 8;
  FragU f;
  f.h[0] = *(const v8h*)(p);
  f.h[1] = *(const v8h*)(p + 16);
  return f.v;
}

// ---------------------------------------------------------------------------
// K0: convert 4 weight matrices (256x256 f32) to f16, concatenated.
// ---------------------------------------------------------------------------
__global__ __launch_bounds__(256) void wcvt_kernel(
    const float* __restrict__ wq, const float* __restrict__ wk,
    const float* __restrict__ wv, const float* __restrict__ wo,
    _Float16* __restrict__ w16) {
  int i = blockIdx.x * 256 + threadIdx.x;  // 0 .. 262143
  int sel = i >> 16;
  int off = i & 65535;
  float val = (sel == 0) ? wq[off] : (sel == 1) ? wk[off]
            : (sel == 2) ? wv[off] : wo[off];
  w16[i] = (_Float16)val;
}

// ---------------------------------------------------------------------------
// K1: GroupNorm, 1 block per (b, group). Writes xn as f16 tokens (b, n, c).
// ---------------------------------------------------------------------------
__global__ __launch_bounds__(256) void groupnorm_kernel(
    const float* __restrict__ x, const float* __restrict__ gamma,
    const float* __restrict__ beta, _Float16* __restrict__ xn) {
  __shared__ float red[256];
  __shared__ float s_mean, s_rstd;
  const int b = blockIdx.x >> 5;
  const int g = blockIdx.x & 31;
  const int tid = threadIdx.x;
  const float* xg = x + ((size_t)b * 256 + g * 8) * 4096;

  float sum = 0.f, sumsq = 0.f;
  for (int i = tid; i < 8 * 4096; i += 256) {
    float v = xg[i];
    sum += v;
    sumsq += v * v;
  }
  red[tid] = sum;
  __syncthreads();
  for (int s = 128; s > 0; s >>= 1) {
    if (tid < s) red[tid] += red[tid + s];
    __syncthreads();
  }
  if (tid == 0) s_mean = red[0] * (1.0f / 32768.0f);
  __syncthreads();
  red[tid] = sumsq;
  __syncthreads();
  for (int s = 128; s > 0; s >>= 1) {
    if (tid < s) red[tid] += red[tid + s];
    __syncthreads();
  }
  if (tid == 0) {
    float mean = s_mean;
    float var = red[0] * (1.0f / 32768.0f) - mean * mean;
    s_rstd = rsqrtf(var + 1e-5f);
  }
  __syncthreads();
  const float mean = s_mean, rstd = s_rstd;
  for (int i = tid; i < 8 * 4096; i += 256) {
    int cl = i >> 12;
    int n = i & 4095;
    int c = g * 8 + cl;
    float val = (xg[i] - mean) * rstd * gamma[c] + beta[c];
    xn[((size_t)b * 4096 + n) * 256 + c] = (_Float16)val;
  }
}

// ---------------------------------------------------------------------------
// Shared tile stager for the projection GEMMs (A 128x32, Bt 64x32, ld=40).
// ---------------------------------------------------------------------------
__device__ inline void stage_gemm_tiles(const _Float16* __restrict__ xb,
                                        const _Float16* __restrict__ w16,
                                        _Float16* At, _Float16* Bt, int m0,
                                        int n0, int k0, int tid) {
#pragma unroll
  for (int i = 0; i < 2; ++i) {  // A: 512 x 16B granules
    int G = tid + i * 256;
    int row = G >> 2, part = (G & 3) * 8;
    async_copy16B(xb + (size_t)(m0 + row) * 256 + k0 + part,
                  At + row * 40 + part);
  }
  {  // Bt[n][k] = W[n0+n][k0+k]: 256 x 16B granules
    int n = tid >> 2, part = (tid & 3) * 8;
    async_copy16B(w16 + (size_t)(n0 + n) * 256 + k0 + part,
                  Bt + n * 40 + part);
  }
}

// ---------------------------------------------------------------------------
// K2: projection GEMM: out = xn(4096x256) @ W^T per batch. Double-buffered
// async LDS tiles; 8 waves x (32x32) wave tiles. Output head-split f16.
// ---------------------------------------------------------------------------
__global__ __launch_bounds__(256) void qkv_gemm_kernel(
    const _Float16* __restrict__ xn, const _Float16* __restrict__ w16,
    _Float16* __restrict__ out, float out_scale) {
  __shared__ _Float16 At[2][128 * 40];
  __shared__ _Float16 Bt[2][64 * 40];
  const int tid = threadIdx.x;
  const int lane = tid & 31;
  const int wv = tid >> 5;
  const int wm = wv & 3;
  const int wn = wv >> 2;
  const int m0 = blockIdx.x * 128;
  const int n0 = blockIdx.y * 64;
  const int b = blockIdx.z;
  const _Float16* xb = xn + (size_t)b * 4096 * 256;

  stage_gemm_tiles(xb, w16, At[0], Bt[0], m0, n0, 0, tid);
  wait_async_copies();
  __syncthreads();

  v8f acc[2][2] = {};
  for (int kt = 0; kt < 8; ++kt) {
    const int cur = kt & 1;
    if (kt + 1 < 8)
      stage_gemm_tiles(xb, w16, At[cur ^ 1], Bt[cur ^ 1], m0, n0,
                       (kt + 1) * 32, tid);
    v16h af[2], bf[2];
    af[0] = load_frag_a(&At[cur][(wm * 32) * 40], 40, lane);
    af[1] = load_frag_a(&At[cur][(wm * 32 + 16) * 40], 40, lane);
    bf[0] = load_frag_bt(&Bt[cur][(wn * 32) * 40], 40, lane);
    bf[1] = load_frag_bt(&Bt[cur][(wn * 32 + 16) * 40], 40, lane);
#pragma unroll
    for (int i = 0; i < 2; ++i)
#pragma unroll
      for (int j = 0; j < 2; ++j) acc[i][j] = wmma_f16(af[i], bf[j], acc[i][j]);
    wait_async_copies();
    __syncthreads();
  }

  const int hh = lane >> 4;
  const int ln = lane & 15;
#pragma unroll
  for (int i = 0; i < 2; ++i) {
#pragma unroll
    for (int j = 0; j < 2; ++j) {
      int ncol = n0 + wn * 32 + j * 16 + ln;
      int head = ncol >> 6;
      int d = ncol & 63;
      _Float16* op = out + ((size_t)(b * 4 + head) * 4096) * 64 + d;
#pragma unroll
      for (int r = 0; r < 8; ++r) {
        int mrow = m0 + wm * 32 + i * 16 + hh * 8 + r;
        op[(size_t)mrow * 64] = (_Float16)(acc[i][j][r] * out_scale);
      }
    }
  }
}

// ---------------------------------------------------------------------------
// K3: flash attention. grid (32, B*heads). 8 waves/block, 16 q-rows/wave.
// Double-buffered 64-key chunks: K via async global->LDS DMA, V prefetched
// into registers at iteration top and transposed into LDS after compute.
// ---------------------------------------------------------------------------
#define FA_LD 72
__global__ __launch_bounds__(256) void flash_attn_kernel(
    const _Float16* __restrict__ q, const _Float16* __restrict__ k,
    const _Float16* __restrict__ v, _Float16* __restrict__ ctx) {
  __shared__ _Float16 Kc[2][64 * FA_LD];  // K chunk raw: Kc[key][d]
  __shared__ _Float16 Vt[2][64 * FA_LD];  // V chunk transposed: Vt[d][key]
  __shared__ _Float16 Pb[8 * 16 * 64];    // per-wave P staging
  const int tid = threadIdx.x;
  const int lane = tid & 31;
  const int wv = tid >> 5;
  const int bh = blockIdx.y;  // b*4 + head
  const int q0 = blockIdx.x * 128 + wv * 16;
  const _Float16* qh = q + ((size_t)bh * 4096 + q0) * 64;
  const _Float16* kh = k + (size_t)bh * 4096 * 64;
  const _Float16* vh = v + (size_t)bh * 4096 * 64;

  const int vkey = (tid * 16) >> 6;  // 0..63 (key this thread transposes)
  const int vd0 = (tid * 16) & 63;   // 0/16/32/48

  v16h qa[2];
  qa[0] = load_frag_a(qh, 64, lane);       // d 0..31
  qa[1] = load_frag_a(qh + 32, 64, lane);  // d 32..63

  v8f o[4] = {};
  float mrow[8], lrow[8];
#pragma unroll
  for (int r = 0; r < 8; ++r) {
    mrow[r] = -1e30f;
    lrow[r] = 0.f;
  }

  _Float16* Pw = Pb + wv * (16 * 64);
  const int hh = lane >> 4;
  const int ln = lane & 15;

  // prologue: stage chunk 0 into buffer 0
  {
#pragma unroll
    for (int i = 0; i < 2; ++i) {
      int G = tid + i * 256;
      int r = G >> 3, gg = (G & 7) * 8;
      async_copy16B(kh + (size_t)r * 64 + gg, &Kc[0][r * FA_LD + gg]);
    }
    const _Float16* vs = vh + (size_t)vkey * 64 + vd0;
    v8h va = *(const v8h*)vs, vb = *(const v8h*)(vs + 8);
#pragma unroll
    for (int i = 0; i < 8; ++i) Vt[0][(vd0 + i) * FA_LD + vkey] = va[i];
#pragma unroll
    for (int i = 0; i < 8; ++i) Vt[0][(vd0 + 8 + i) * FA_LD + vkey] = vb[i];
  }
  wait_async_copies();
  __syncthreads();

  for (int j = 0; j < 64; ++j) {
    const int cur = j & 1;
    const int nxt = cur ^ 1;
    const bool havenext = (j + 1) < 64;
    v8h va = {}, vb = {};
    if (havenext) {  // issue next chunk's K DMA + V register prefetch
      const int kbn = (j + 1) * 64;
#pragma unroll
      for (int i = 0; i < 2; ++i) {
        int G = tid + i * 256;
        int r = G >> 3, gg = (G & 7) * 8;
        async_copy16B(kh + (size_t)(kbn + r) * 64 + gg,
                      &Kc[nxt][r * FA_LD + gg]);
      }
      const _Float16* vs = vh + (size_t)(kbn + vkey) * 64 + vd0;
      va = *(const v8h*)vs;
      vb = *(const v8h*)(vs + 8);
    }

    // S = Q * K^T (16 x 64 per wave); Q pre-scaled by 1/sqrt(d)
    v8f s[4] = {};
#pragma unroll
    for (int tj = 0; tj < 4; ++tj)
#pragma unroll
      for (int ks = 0; ks < 2; ++ks) {
        v16h bf =
            load_frag_bt(&Kc[cur][(tj * 16) * FA_LD + ks * 32], FA_LD, lane);
        s[tj] = wmma_f16(qa[ks], bf, s[tj]);
      }

    // Online softmax: accumulator row r lives in one VGPR slot across a
    // 16-lane half; xor-shuffles 1/2/4/8 reduce within the half.
    float mnew[8], alpha[8];
#pragma unroll
    for (int r = 0; r < 8; ++r) {
      float mx = fmaxf(fmaxf(s[0][r], s[1][r]), fmaxf(s[2][r], s[3][r]));
      mx = fmaxf(mx, __shfl_xor(mx, 1));
      mx = fmaxf(mx, __shfl_xor(mx, 2));
      mx = fmaxf(mx, __shfl_xor(mx, 4));
      mx = fmaxf(mx, __shfl_xor(mx, 8));
      mnew[r] = fmaxf(mrow[r], mx);
      alpha[r] = __expf(mrow[r] - mnew[r]);
      mrow[r] = mnew[r];
    }
#pragma unroll
    for (int r = 0; r < 8; ++r) {
      float rs = 0.f;
#pragma unroll
      for (int tj = 0; tj < 4; ++tj) {
        float pv = __expf(s[tj][r] - mnew[r]);
        s[tj][r] = pv;
        rs += pv;
      }
      rs += __shfl_xor(rs, 1);
      rs += __shfl_xor(rs, 2);
      rs += __shfl_xor(rs, 4);
      rs += __shfl_xor(rs, 8);
      lrow[r] = lrow[r] * alpha[r] + rs;
#pragma unroll
      for (int tj = 0; tj < 4; ++tj) o[tj][r] *= alpha[r];
    }

    // Stage P (f16) in per-wave LDS; reload in A-fragment order.
#pragma unroll
    for (int tj = 0; tj < 4; ++tj)
#pragma unroll
      for (int r = 0; r < 8; ++r)
        Pw[(hh * 8 + r) * 64 + tj * 16 + ln] = (_Float16)s[tj][r];
    asm volatile("s_wait_dscnt 0x0" ::: "memory");  // same-wave LDS RAW

    v16h pa[2];
    pa[0] = load_frag_a(Pw, 64, lane);
    pa[1] = load_frag_a(Pw + 32, 64, lane);
#pragma unroll
    for (int tj = 0; tj < 4; ++tj)
#pragma unroll
      for (int ks = 0; ks < 2; ++ks) {
        v16h bf =
            load_frag_bt(&Vt[cur][(tj * 16) * FA_LD + ks * 32], FA_LD, lane);
        o[tj] = wmma_f16(pa[ks], bf, o[tj]);
      }

    if (havenext) {  // commit next chunk's V transpose
#pragma unroll
      for (int i = 0; i < 8; ++i) Vt[nxt][(vd0 + i) * FA_LD + vkey] = va[i];
#pragma unroll
      for (int i = 0; i < 8; ++i)
        Vt[nxt][(vd0 + 8 + i) * FA_LD + vkey] = vb[i];
    }
    wait_async_copies();
    __syncthreads();
  }

  // ctx (b, n, c) with c = head*64 + d
  const int b = bh >> 2;
  const int head = bh & 3;
#pragma unroll
  for (int tj = 0; tj < 4; ++tj) {
    int col = head * 64 + tj * 16 + ln;
#pragma unroll
    for (int r = 0; r < 8; ++r) {
      int row = q0 + hh * 8 + r;
      float inv = 1.0f / lrow[r];
      ctx[((size_t)b * 4096 + row) * 256 + col] = (_Float16)(o[tj][r] * inv);
    }
  }
}

// ---------------------------------------------------------------------------
// K4: output projection + residual, f32 epilogue into (b, c, h, w).
// ---------------------------------------------------------------------------
__global__ __launch_bounds__(256) void out_proj_kernel(
    const _Float16* __restrict__ ctx, const _Float16* __restrict__ w16,
    const float* __restrict__ xres, float* __restrict__ out) {
  __shared__ _Float16 At[2][128 * 40];
  __shared__ _Float16 Bt[2][64 * 40];
  const int tid = threadIdx.x;
  const int lane = tid & 31;
  const int wv = tid >> 5;
  const int wm = wv & 3;
  const int wn = wv >> 2;
  const int m0 = blockIdx.x * 128;
  const int n0 = blockIdx.y * 64;
  const int b = blockIdx.z;
  const _Float16* cb = ctx + (size_t)b * 4096 * 256;

  stage_gemm_tiles(cb, w16, At[0], Bt[0], m0, n0, 0, tid);
  wait_async_copies();
  __syncthreads();

  v8f acc[2][2] = {};
  for (int kt = 0; kt < 8; ++kt) {
    const int cur = kt & 1;
    if (kt + 1 < 8)
      stage_gemm_tiles(cb, w16, At[cur ^ 1], Bt[cur ^ 1], m0, n0,
                       (kt + 1) * 32, tid);
    v16h af[2], bf[2];
    af[0] = load_frag_a(&At[cur][(wm * 32) * 40], 40, lane);
    af[1] = load_frag_a(&At[cur][(wm * 32 + 16) * 40], 40, lane);
    bf[0] = load_frag_bt(&Bt[cur][(wn * 32) * 40], 40, lane);
    bf[1] = load_frag_bt(&Bt[cur][(wn * 32 + 16) * 40], 40, lane);
#pragma unroll
    for (int i = 0; i < 2; ++i)
#pragma unroll
      for (int j = 0; j < 2; ++j) acc[i][j] = wmma_f16(af[i], bf[j], acc[i][j]);
    wait_async_copies();
    __syncthreads();
  }

  const int hh = lane >> 4;
  const int ln = lane & 15;
#pragma unroll
  for (int i = 0; i < 2; ++i) {
#pragma unroll
    for (int j = 0; j < 2; ++j) {
      int ncol = n0 + wn * 32 + j * 16 + ln;
#pragma unroll
      for (int r = 0; r < 8; ++r) {
        int mrow = m0 + wm * 32 + i * 16 + hh * 8 + r;
        size_t idx = ((size_t)b * 256 + ncol) * 4096 + mrow;
        out[idx] = acc[i][j][r] + xres[idx];
      }
    }
  }
}

// ---------------------------------------------------------------------------
// Host launch
// ---------------------------------------------------------------------------
extern "C" void kernel_launch(void* const* d_in, const int* in_sizes, int n_in,
                              void* d_out, int out_size, void* d_ws,
                              size_t ws_size, hipStream_t stream) {
  (void)in_sizes; (void)n_in; (void)out_size; (void)ws_size;
  const float* x     = (const float*)d_in[0];
  const float* gamma = (const float*)d_in[1];
  const float* beta  = (const float*)d_in[2];
  const float* wq    = (const float*)d_in[3];
  const float* wk    = (const float*)d_in[4];
  const float* wv    = (const float*)d_in[5];
  const float* wo    = (const float*)d_in[6];
  float* out = (float*)d_out;

  char* ws = (char*)d_ws;
  _Float16* xn  = (_Float16*)(ws);                      // 8 MB (reused as ctx)
  _Float16* q16 = (_Float16*)(ws + ((size_t)8 << 20));  // 8 MB
  _Float16* k16 = (_Float16*)(ws + ((size_t)16 << 20)); // 8 MB
  _Float16* v16 = (_Float16*)(ws + ((size_t)24 << 20)); // 8 MB
  _Float16* w16 = (_Float16*)(ws + ((size_t)32 << 20)); // 512 KB
  _Float16* wq16 = w16;
  _Float16* wk16 = w16 + 65536;
  _Float16* wv16 = w16 + 2 * 65536;
  _Float16* wo16 = w16 + 3 * 65536;

  wcvt_kernel<<<1024, 256, 0, stream>>>(wq, wk, wv, wo, w16);
  groupnorm_kernel<<<128, 256, 0, stream>>>(x, gamma, beta, xn);

  dim3 gg(32, 4, 4);  // 4096/128 m-blocks, 256/64 n-blocks, B batches
  qkv_gemm_kernel<<<gg, 256, 0, stream>>>(xn, wq16, q16, 0.125f);
  qkv_gemm_kernel<<<gg, 256, 0, stream>>>(xn, wk16, k16, 1.0f);
  qkv_gemm_kernel<<<gg, 256, 0, stream>>>(xn, wv16, v16, 1.0f);

  flash_attn_kernel<<<dim3(32, 16), 256, 0, stream>>>(q16, k16, v16, xn);

  out_proj_kernel<<<gg, 256, 0, stream>>>(xn, wo16, x, out);
}
